// Model_83442624627142
// MI455X (gfx1250) — matrix operand order repaired and checked by reading, not compiled, not run on hardware
//
#include <hip/hip_runtime.h>
#include <hip/hip_bf16.h>
#include <math.h>

typedef __bf16 bf16;
typedef bf16  v16bf  __attribute__((ext_vector_type(16)));
typedef bf16  bf16x8 __attribute__((ext_vector_type(8)));
typedef bf16  bf16x4 __attribute__((ext_vector_type(4)));
typedef float v8f    __attribute__((ext_vector_type(8)));
typedef unsigned int u32x4 __attribute__((ext_vector_type(4)));
typedef int   i32x4  __attribute__((ext_vector_type(4)));
typedef int   i32x8  __attribute__((ext_vector_type(8)));

#define V_DIM 32000
#define E_DIM 1024
#define H_DIM 1024
#define T_DIM 2048
#define G4H   (4*H_DIM)

// ---------------------------------------------------------------- utilities

__global__ void init_barrier_kernel(unsigned* p) {
    if (threadIdx.x < 2) p[threadIdx.x] = 0u;
}

__global__ void f32_to_bf16_kernel(const float* __restrict__ src,
                                   bf16* __restrict__ dst, int n4) {
    int i = blockIdx.x * blockDim.x + threadIdx.x;
    if (i < n4) {
        const float4 v = ((const float4*)src)[i];
        bf16x4 o;
        o[0] = (bf16)v.x; o[1] = (bf16)v.y; o[2] = (bf16)v.z; o[3] = (bf16)v.w;
        ((bf16x4*)dst)[i] = o;
    }
}

__global__ void gather_embed_kernel(const int* __restrict__ seq,
                                    const float* __restrict__ emb,
                                    bf16* __restrict__ x) {
    const int e4 = E_DIM >> 2;
    int i = blockIdx.x * blockDim.x + threadIdx.x;   // over T*E/4
    int t = i / e4, c = i - t * e4;
    if (t < T_DIM) {
        const float4 v = ((const float4*)(emb + (size_t)seq[t] * E_DIM))[c];
        bf16x4 o;
        o[0] = (bf16)v.x; o[1] = (bf16)v.y; o[2] = (bf16)v.z; o[3] = (bf16)v.w;
        ((bf16x4*)(x + (size_t)t * E_DIM))[c] = o;
    }
}

// ----------------------------------------------------------- TDM tile load
// One descriptor moves a 128-row x 32-element bf16 tile (row pitch = `pitch`
// elements) from global into LDS. pad_enable inserts 4 DWORDs of padding
// after every 16 DWORDs (= one 32-element row), reproducing the 40-half LDS
// row stride the WMMA fragment loaders expect. Tensor dims == tile dims
// (tile origin is the global address), so no OOB cases arise.

__device__ __forceinline__ void tdm_load_tile(const bf16* __restrict__ gptr,
                                              unsigned lds_off, int pitch) {
    const unsigned long long ga = (unsigned long long)(const void*)gptr;
    u32x4 g0;
    g0[0] = 1u;                                    // count=1, user descriptor
    g0[1] = lds_off;                               // LDS byte offset
    g0[2] = (unsigned)ga;                          // global_addr[31:0]
    g0[3] = (unsigned)((ga >> 32) & 0x01ffffffu)   // global_addr[56:32]
            | (2u << 30);                          // type = 2 ("image")
    i32x8 g1;
    g1[0] = (int)((1u << 16)                       // data_size = 2 bytes
                  | (1u << 20)                     // pad_enable
                  | (3u << 22)                     // pad_interval = 16 DW
                  | (3u << 25));                   // pad_amount   = 4 DW
    g1[1] = (int)(32u << 16);                      // tensor_dim0 = 32
    g1[2] = (int)(128u << 16);                     // tensor_dim1 = 128
    g1[3] = (int)(32u << 16);                      // tile_dim0 = 32
    g1[4] = 128;                                   // tile_dim1 = 128
    g1[5] = pitch;                                 // tensor_dim0_stride
    g1[6] = 0;
    g1[7] = 0;
    i32x4 z4 = {0, 0, 0, 0};
    i32x8 z8 = {0, 0, 0, 0, 0, 0, 0, 0};
    __builtin_amdgcn_tensor_load_to_lds(g0, g1, z4, z4, z8, 0);
}

// ---------------------------------------------------------------- WMMA GEMM
// C[M,N] (f32) = A[M,K] (bf16, row-major) * B[N,K]^T (bf16, row-major "W")
//               + bias (NB compile-time vectors)
// Block tile 128x128, 8 waves, each wave 32x64 (2x4 frags), K-step 32.
// Tiles are DMA'd into double-buffered LDS by the Tensor Data Mover:
// wave 0 feeds A, wave 1 feeds B, TENSORcnt + s_barrier publish each tile
// while the 8 WMMAs of the previous tile execute.

__device__ __forceinline__ v16bf frag_a(const bf16* __restrict__ tile,
                                        int localM, int lane) {
    // 16-bit A 16x32 layout: lanes 0-15 hold K {0..7,16..23}, lanes 16-31
    // hold K {8..15,24..31}; VGPR j packs K=2j,2j+1.
    const int r = lane & 15, hi = lane >> 4;
    const bf16* p = tile + (localM + r) * 40 + hi * 8;
    bf16x4 c0 = *(const bf16x4*)(p);
    bf16x4 c1 = *(const bf16x4*)(p + 4);
    bf16x4 c2 = *(const bf16x4*)(p + 16);
    bf16x4 c3 = *(const bf16x4*)(p + 20);
    v16bf a;
#pragma unroll
    for (int i = 0; i < 4; ++i) {
        a[i] = c0[i]; a[4 + i] = c1[i]; a[8 + i] = c2[i]; a[12 + i] = c3[i];
    }
    return a;
}

__device__ __forceinline__ v16bf frag_b(const bf16* __restrict__ tile,
                                        int localN, int lane) {
    // 16-bit B 32x16 layout: lanes 0-15 -> K {0..7,16..23}, lanes 16-31 -> K {8..15,24..31}.
    const int c = lane & 15, hi = lane >> 4;
    const bf16* p = tile + (localN + c) * 40 + hi * 8;
    bf16x4 c0 = *(const bf16x4*)(p);
    bf16x4 c1 = *(const bf16x4*)(p + 4);
    bf16x4 c2 = *(const bf16x4*)(p + 16);
    bf16x4 c3 = *(const bf16x4*)(p + 20);
    v16bf b;
#pragma unroll
    for (int i = 0; i < 4; ++i) {
        b[i] = c0[i]; b[4 + i] = c1[i]; b[8 + i] = c2[i]; b[12 + i] = c3[i];
    }
    return b;
}

template <int NB>
__global__ __launch_bounds__(256)
void gemm_bf16_kernel(const bf16* __restrict__ A, const bf16* __restrict__ B,
                      const float* __restrict__ bias0,
                      const float* __restrict__ bias1,
                      float* __restrict__ C, int K, int ldc) {
    __shared__ bf16 As[2][128 * 40];
    __shared__ bf16 Bs[2][128 * 40];

    const int tid  = threadIdx.x;
    const int lane = tid & 31, wave = tid >> 5;
    const int wm = wave & 3, wn = wave >> 2;           // 4 (M) x 2 (N) waves
    const int m0 = blockIdx.y * 128, n0 = blockIdx.x * 128;

    const bf16* arow0 = A + (size_t)m0 * K;            // block tile origins
    const bf16* brow0 = B + (size_t)n0 * K;
    const unsigned aoff[2] = { (unsigned)(uintptr_t)&As[0][0],
                               (unsigned)(uintptr_t)&As[1][0] };
    const unsigned boff[2] = { (unsigned)(uintptr_t)&Bs[0][0],
                               (unsigned)(uintptr_t)&Bs[1][0] };

    v8f acc[2][4];
#pragma unroll
    for (int i = 0; i < 2; ++i)
#pragma unroll
        for (int j = 0; j < 4; ++j) acc[i][j] = {};

    // Prologue: DMA tile 0 into buffer 0 (wave 0 -> A, wave 1 -> B).
    if (wave == 0) tdm_load_tile(arow0, aoff[0], K);
    if (wave == 1) tdm_load_tile(brow0, boff[0], K);

    const int nk = K >> 5;
    for (int i = 0; i < nk; ++i) {
        if (wave <= 1) __builtin_amdgcn_s_wait_tensorcnt(0);
        __syncthreads();              // tile i visible; buf (i+1)&1 drained
        const int cur = i & 1;

        if (i + 1 < nk) {             // DMA next tile while WMMAs run
            const int k0n = (i + 1) * 32;
            const int nxt = cur ^ 1;
            if (wave == 0) tdm_load_tile(arow0 + k0n, aoff[nxt], K);
            if (wave == 1) tdm_load_tile(brow0 + k0n, boff[nxt], K);
        }

        v16bf af[2], bfr[4];
#pragma unroll
        for (int mt = 0; mt < 2; ++mt)
            af[mt] = frag_a(&As[cur][0], wm * 32 + mt * 16, lane);
#pragma unroll
        for (int nt = 0; nt < 4; ++nt)
            bfr[nt] = frag_b(&Bs[cur][0], wn * 64 + nt * 16, lane);
#pragma unroll
        for (int mt = 0; mt < 2; ++mt)
#pragma unroll
            for (int nt = 0; nt < 4; ++nt)
                acc[mt][nt] = __builtin_amdgcn_wmma_f32_16x16x32_bf16(
                    false, af[mt], false, bfr[nt], (short)0, acc[mt][nt],
                    false, false);
    }

    const int hi = lane >> 4, cl = lane & 15;
#pragma unroll
    for (int mt = 0; mt < 2; ++mt)
#pragma unroll
        for (int nt = 0; nt < 4; ++nt) {
            const int col = n0 + wn * 64 + nt * 16 + cl;
            float bb = 0.f;
            if (NB >= 1) bb += bias0[col];
            if (NB >= 2) bb += bias1[col];
            const int rbase = m0 + wm * 32 + mt * 16 + hi * 8;
#pragma unroll
            for (int r = 0; r < 8; ++r)
                C[(size_t)(rbase + r) * ldc + col] = acc[mt][nt][r] + bb;
        }
}

// ---------------------------------------------------------------- LSTM scan
// 32 persistent WGs x 256 threads. WG g owns h-indices [g*32, g*32+32) i.e.
// gate rows {q*1024 + g*32 + jj}. Its 128x1024 W_hh slice lives in LDS as
// bf16 (256 KB of the 320 KB WGP LDS). h crosses WGs via global f32 + a
// sense-reversing device barrier each step; c never leaves the WGP.

__device__ __forceinline__ float sigf(float x) { return 1.f / (1.f + __expf(-x)); }

__global__ __launch_bounds__(256)
void lstm_kernel(const float* __restrict__ xg, const float* __restrict__ Whh,
                 const float* __restrict__ h0, const float* __restrict__ c0,
                 float* __restrict__ hglob, bf16* __restrict__ hs,
                 float* __restrict__ hT, float* __restrict__ cT,
                 unsigned* __restrict__ bar_cnt, unsigned* __restrict__ bar_gen,
                 int nblocks) {
    extern __shared__ char smem[];
    bf16*  Ws     = (bf16*)smem;                                   // 128*1024
    float* h_l    = (float*)(smem + 128 * 1024 * sizeof(bf16));    // 1024
    float* gatesl = h_l + 1024;                                    // 128
    float* c_l    = gatesl + 128;                                  // 32

    const int tid = threadIdx.x;
    const int g   = blockIdx.x;

    // Preload + quantize the W_hh slice into LDS (coalesced float4 reads).
    for (int lr = 0; lr < 128; ++lr) {
        const int gr = (lr >> 5) * 1024 + g * 32 + (lr & 31);
        const float4 v = ((const float4*)(Whh + (size_t)gr * H_DIM))[tid];
        bf16x4 o;
        o[0] = (bf16)v.x; o[1] = (bf16)v.y; o[2] = (bf16)v.z; o[3] = (bf16)v.w;
        ((bf16x4*)(Ws + lr * 1024))[tid] = o;
    }
    if (tid < 32) c_l[tid] = c0[g * 32 + tid];

    const int row = tid >> 1, kh = tid & 1;            // 2 lanes per gate row
    const int grow = (row >> 5) * 1024 + g * 32 + (row & 31);
    const bf16* wrow = Ws + row * 1024 + kh * 512;

    unsigned expected = 0;
    for (int t = 0; t < T_DIM; ++t) {
        const float* hsrc = (t == 0) ? h0 : hglob;
        ((float4*)h_l)[tid] = ((const float4*)hsrc)[tid];
        __syncthreads();

        float acc = (kh == 0) ? xg[(size_t)t * G4H + grow] : 0.f;
        const float* hp = h_l + kh * 512;
#pragma unroll 8
        for (int k = 0; k < 512; k += 4) {
            bf16x4 w  = *(const bf16x4*)(wrow + k);
            float4 h4 = *(const float4*)(hp + k);
            acc += (float)w[0] * h4.x; acc += (float)w[1] * h4.y;
            acc += (float)w[2] * h4.z; acc += (float)w[3] * h4.w;
        }
        acc += __shfl_xor(acc, 1);
        if (kh == 0) gatesl[row] = acc;
        __syncthreads();

        if (tid < 32) {
            const int jj = tid;
            const float gi = sigf(gatesl[jj]);
            const float gf = sigf(gatesl[32 + jj]);
            const float gg = tanhf(gatesl[64 + jj]);
            const float go = sigf(gatesl[96 + jj]);
            const float c  = gf * c_l[jj] + gi * gg;
            const float h  = go * tanhf(c);
            c_l[jj] = c;
            const int j = g * 32 + jj;
            hglob[j] = h;
            hs[(size_t)t * H_DIM + j] = (bf16)h;
            if (t == T_DIM - 1) { hT[j] = h; cT[j] = c; }
        }

        // Device-wide sense-reversing barrier (s_sleep spin).
        __syncthreads();
        ++expected;
        if (tid == 0) {
            __threadfence();
            unsigned a = atomicAdd(bar_cnt, 1u);
            if (a == (unsigned)nblocks - 1u) {
                atomicSub(bar_cnt, (unsigned)nblocks);
                __threadfence();
                atomicAdd(bar_gen, 1u);
            }
            while (atomicAdd(bar_gen, 0u) < expected)
                __builtin_amdgcn_s_sleep(2);
            __threadfence();
        }
        __syncthreads();
    }
}

// ---------------------------------------------------------------- softmax
// One block per row; the 125 KB score row is staged in LDS so HBM sees
// exactly one read + one write per element.

__global__ __launch_bounds__(256)
void logsoftmax_kernel(float* __restrict__ S) {
    extern __shared__ char smem2[];
    float* buf = (float*)smem2;
    __shared__ float red[8];
    const int tid = threadIdx.x, lane = tid & 31, wave = tid >> 5;
    float* row = S + (size_t)blockIdx.x * V_DIM;
    float4* b4 = (float4*)buf;
    const float4* r4 = (const float4*)row;
    const int n4 = V_DIM / 4;

    float m = -3.4e38f;
    for (int i = tid; i < n4; i += 256) {
        float4 v = r4[i]; b4[i] = v;
        m = fmaxf(m, fmaxf(fmaxf(v.x, v.y), fmaxf(v.z, v.w)));
    }
#pragma unroll
    for (int o = 16; o > 0; o >>= 1) m = fmaxf(m, __shfl_xor(m, o));
    if (lane == 0) red[wave] = m;
    __syncthreads();
    m = red[0];
#pragma unroll
    for (int w = 1; w < 8; ++w) m = fmaxf(m, red[w]);

    float s = 0.f;
    for (int i = tid; i < n4; i += 256) {
        float4 v = b4[i];
        s += __expf(v.x - m) + __expf(v.y - m) + __expf(v.z - m) + __expf(v.w - m);
    }
#pragma unroll
    for (int o = 16; o > 0; o >>= 1) s += __shfl_xor(s, o);
    __syncthreads();
    if (lane == 0) red[wave] = s;
    __syncthreads();
    s = 0.f;
#pragma unroll
    for (int w = 0; w < 8; ++w) s += red[w];

    const float lse = m + __logf(s);
    for (int i = tid; i < n4; i += 256) {
        float4 v = b4[i];
        v.x -= lse; v.y -= lse; v.z -= lse; v.w -= lse;
        ((float4*)row)[i] = v;
    }
}

// ---------------------------------------------------------------- launcher

extern "C" void kernel_launch(void* const* d_in, const int* in_sizes, int n_in,
                              void* d_out, int out_size, void* d_ws,
                              size_t ws_size, hipStream_t stream) {
    (void)in_sizes; (void)n_in; (void)out_size; (void)ws_size;
    const int*   seq  = (const int*)  d_in[0];
    const float* h0   = (const float*)d_in[1];
    const float* c0   = (const float*)d_in[2];
    const float* emb  = (const float*)d_in[3];
    const float* Wih  = (const float*)d_in[4];
    const float* Whh  = (const float*)d_in[5];
    const float* bih  = (const float*)d_in[6];
    const float* bhh  = (const float*)d_in[7];
    const float* Wout = (const float*)d_in[8];
    const float* bout = (const float*)d_in[9];

    char* ws = (char*)d_ws;
    size_t off = 0;
    auto take = [&](size_t bytes) -> char* {
        char* p = ws + off;
        off = (off + bytes + 255) & ~(size_t)255;
        return p;
    };
    bf16*     x_bf    = (bf16*)    take((size_t)T_DIM * E_DIM * 2);
    bf16*     Wih_bf  = (bf16*)    take((size_t)G4H   * E_DIM * 2);
    bf16*     Wout_bf = (bf16*)    take((size_t)V_DIM * H_DIM * 2);
    float*    xg      = (float*)   take((size_t)T_DIM * G4H   * 4);
    bf16*     hs_bf   = (bf16*)    take((size_t)T_DIM * H_DIM * 2);
    float*    hglob   = (float*)   take((size_t)H_DIM * 4);
    unsigned* bar     = (unsigned*)take(256);

    float* logp = (float*)d_out;
    float* hT   = logp + (size_t)T_DIM * V_DIM;
    float* cT   = hT + H_DIM;

    init_barrier_kernel<<<1, 64, 0, stream>>>(bar);
    gather_embed_kernel<<<(T_DIM * E_DIM / 4 + 255) / 256, 256, 0, stream>>>(
        seq, emb, x_bf);
    f32_to_bf16_kernel<<<(G4H * E_DIM / 4 + 255) / 256, 256, 0, stream>>>(
        Wih, Wih_bf, G4H * E_DIM / 4);
    f32_to_bf16_kernel<<<(V_DIM * H_DIM / 4 + 255) / 256, 256, 0, stream>>>(
        Wout, Wout_bf, V_DIM * H_DIM / 4);

    dim3 gx(G4H / 128, T_DIM / 128);
    gemm_bf16_kernel<2><<<gx, 256, 0, stream>>>(x_bf, Wih_bf, bih, bhh, xg,
                                                E_DIM, G4H);

    const size_t lstm_lds = 128 * 1024 * 2 + 1024 * 4 + 128 * 4 + 32 * 4;
    lstm_kernel<<<32, 256, lstm_lds, stream>>>(xg, Whh, h0, c0, hglob, hs_bf,
                                               hT, cT, bar, bar + 1, 32);

    dim3 gs(V_DIM / 128, T_DIM / 128);
    gemm_bf16_kernel<1><<<gs, 256, 0, stream>>>(hs_bf, Wout_bf, bout, nullptr,
                                                logp, H_DIM, V_DIM);

    logsoftmax_kernel<<<T_DIM, 256, (size_t)V_DIM * 4, stream>>>(logp);
}
